// EchoStateLayer_6150393168395
// MI455X (gfx1250) — matrix operand order repaired and checked by reading, not compiled, not run on hardware
//
#include <hip/hip_runtime.h>
#include <cmath>

typedef __attribute__((ext_vector_type(16))) _Float16 v16h;
typedef __attribute__((ext_vector_type(8)))  _Float16 v8h;
typedef __attribute__((ext_vector_type(8)))  float    v8f;
typedef __attribute__((ext_vector_type(4)))  unsigned u32x4;
typedef __attribute__((ext_vector_type(8)))  int      i32x8;
typedef __attribute__((ext_vector_type(4)))  int      i32x4;

union V16 { v16h v; v8h h[2]; };

#define B_  32
#define T_  512
#define D_  256
#define U_  1024

static __device__ __forceinline__ v8f wmma_f16(const V16& a, const V16& b, v8f c) {
    return __builtin_amdgcn_wmma_f32_16x16x32_f16(false, a.v, false, b.v,
                                                  (short)0, c, false, false);
}

// ---------------------------------------------------------------------------
// Kernel 0: prep.
//  - inputs f32 -> f16                    (a16,   [B*T, D])
//  - W_in   f32 -> f16 transposed         (bt16,  [U, D])
//  - W_res  f32 -> f16, per-block slices in WMMA-fragment order (wres16, 2 MB)
//  - zero h ping buffer (h_0 = 0) and the grid-barrier counter
// ---------------------------------------------------------------------------
__global__ void esn_prep(const float* __restrict__ x,
                         const float* __restrict__ win,
                         const float* __restrict__ wres,
                         _Float16* __restrict__ a16,
                         _Float16* __restrict__ bt16,
                         _Float16* __restrict__ wres16,
                         _Float16* __restrict__ h,
                         unsigned* __restrict__ counter)
{
    long tid = (long)blockIdx.x * blockDim.x + threadIdx.x;
    long nth = (long)gridDim.x * blockDim.x;
    if (tid == 0) *counter = 0u;

    const long NA = (long)B_ * T_ * D_;
    for (long i = tid; i < NA; i += nth) a16[i] = (_Float16)x[i];

    const long NB = (long)D_ * U_;
    for (long i = tid; i < NB; i += nth) {
        long d = i / U_, u = i % U_;            // coalesced read over u
        bt16[u * D_ + d] = (_Float16)win[i];
    }

    // W_res fragment-sliced: 32 blocks x (32 cols x 1024 k) contiguous 64 KB
    const long NR = (long)U_ * U_;
    for (long i = tid; i < NR; i += nth) {
        int k = (int)(i / U_), n = (int)(i % U_);   // coalesced read over n
        int blk = n >> 5, nc = n & 31;
        int kk = k >> 5,  kr = k & 31;
        int nt = nc >> 4;
        int ln = (nc & 15) + ((kr >> 4) << 4);
        int e  = kr & 15;
        wres16[(long)blk * 32768 + ((kk * 2 + nt) << 9) + (ln << 4) + e] =
            (_Float16)wres[i];
    }

    const long NH = (long)B_ * U_;              // h ping buffer = zeros
    for (long i = tid; i < NH; i += nth) h[i] = (_Float16)0.f;
}

// ---------------------------------------------------------------------------
// Kernel 1: xin = inputs @ W_in + b  ->  straight into d_out (f32).
// One 16x16 tile per wave; K=256 via 8 WMMAs in two independent chains.
// ---------------------------------------------------------------------------
__global__ void __launch_bounds__(256) esn_xin(const _Float16* __restrict__ a16,
                                               const _Float16* __restrict__ bt16,
                                               const float* __restrict__ bias,
                                               float* __restrict__ out)
{
    const int lane  = threadIdx.x & 31;
    const int wave  = threadIdx.x >> 5;
    const int ntile = blockIdx.x;                 // 0..63
    const int mtile = blockIdx.y * 8 + wave;      // 0..1023
    const int rbase = mtile * 16;
    const int nbase = ntile * 16;

    const int kA = (lane >> 4) * 8;
    const int kB = (lane >> 4) * 8;
    const _Float16* aRow = a16  + (long)(rbase + (lane & 15)) * D_ + kA;
    const _Float16* bRow = bt16 + (long)(nbase + (lane & 15)) * D_ + kB;

    v8f acc0 = {}, acc1 = {};
#pragma unroll
    for (int kk = 0; kk < D_ / 32; kk += 2) {
        V16 a0, b0, a1, b1;
        a0.h[0] = *(const v8h*)(aRow + kk * 32);
        a0.h[1] = *(const v8h*)(aRow + kk * 32 + 16);
        b0.h[0] = *(const v8h*)(bRow + kk * 32);
        b0.h[1] = *(const v8h*)(bRow + kk * 32 + 16);
        a1.h[0] = *(const v8h*)(aRow + kk * 32 + 32);
        a1.h[1] = *(const v8h*)(aRow + kk * 32 + 48);
        b1.h[0] = *(const v8h*)(bRow + kk * 32 + 32);
        b1.h[1] = *(const v8h*)(bRow + kk * 32 + 48);
        acc0 = wmma_f16(a0, b0, acc0);
        acc1 = wmma_f16(a1, b1, acc1);
    }
    v8f acc = acc0 + acc1;

    const float bv  = bias[nbase + (lane & 15)];
    const int  mofs = (lane >> 4) * 8;
#pragma unroll
    for (int e = 0; e < 8; ++e)
        out[(long)(rbase + mofs + e) * U_ + nbase + (lane & 15)] = acc[e] + bv;
}

// ---------------------------------------------------------------------------
// Kernel 2: persistent scan. 32 WGs x 128 threads; WG stages its 64 KB W_res
// slice into LDS via the Tensor Data Mover (fallback: vector copy), then runs
// 512 steps of h = tanh(xin + h @ W_res) with dual-chain WMMAs + grid barrier.
// ---------------------------------------------------------------------------
__global__ void __launch_bounds__(128, 1)
esn_scan(const _Float16* __restrict__ wres16, // [32][32768] fragment-ordered
         float* __restrict__ out,             // [B,T,U]: xin in, h_t out
         _Float16* __restrict__ h,            // [2][B][U] f16
         unsigned* __restrict__ counter)
{
    __shared__ _Float16 smem[32 * U_];        // 64 KB W_res slice

    const int tid   = threadIdx.x;
    const int lane  = tid & 31;
    const int wave  = tid >> 5;
    const int mtile = wave & 1;
    const int ntile = wave >> 1;

#if __has_builtin(__builtin_amdgcn_tensor_load_to_lds) && \
    __has_builtin(__builtin_amdgcn_s_wait_tensorcnt)
    // ---- TDM: one 1-D 64 KB tensor DMA, global -> LDS, wave 0 issues it
    if (wave == 0) {
        unsigned ldsOff = (unsigned)(uintptr_t)(&smem[0]);      // low word = LDS offset
        unsigned long long ga =
            (unsigned long long)(uintptr_t)(wres16 + (long)blockIdx.x * 32768);
        u32x4 g0 = { 1u,                       // count=1 (valid descriptor)
                     ldsOff,                   // lds_addr
                     (unsigned)ga,             // global_addr[31:0]
                     (unsigned)(ga >> 32) | (2u << 30) };   // addr[56:32] | type=2
        i32x8 g1 = { (int)(1u << 16),          // data_size = 2 bytes
                     (int)(32768u << 16),      // tensor_dim0 = 32768 (lo16 @bit48)
                     (int)(1u << 16),          // tensor_dim1 = 1
                     (int)(32768u << 16),      // tile_dim0 = 32768
                     0,                        // tile_dim1/2 unused (1-D tile)
                     32768,                    // tensor_dim0_stride[31:0]
                     0, 0 };
        i32x4 gz4 = { 0, 0, 0, 0 };
        i32x8 gz8 = { 0, 0, 0, 0, 0, 0, 0, 0 };
        __builtin_amdgcn_tensor_load_to_lds(g0, g1, gz4, gz4, gz8, 0);
        __builtin_amdgcn_s_wait_tensorcnt(0);
    }
    __syncthreads();
#else
    // ---- fallback: contiguous 64 KB vector copy
    {
        const ulonglong2* src =
            (const ulonglong2*)(wres16 + (long)blockIdx.x * 32768);
        ulonglong2* dst = (ulonglong2*)smem;
        for (int i = tid; i < 4096; i += 128) dst[i] = src[i];
        __syncthreads();
    }
#endif

    const int mrow = mtile * 16 + (lane & 15);
    const int kA   = (lane >> 4) * 16;
    const _Float16* bFrag = smem + ((size_t)ntile << 9) + ((size_t)lane << 4);
    const int ncol = blockIdx.x * 32 + ntile * 16 + (lane & 15);
    const int mofs = mtile * 16 + ((lane >> 4) << 3);

    const unsigned nwg = gridDim.x;

    for (int t = 0; t < T_; ++t) {
        const _Float16* hc = h + ((t & 1) ? B_ * U_ : 0);
        _Float16*       hn = h + ((t & 1) ? 0 : B_ * U_);
        const _Float16* aRow = hc + (long)mrow * U_ + kA;

        __builtin_prefetch(&out[((long)mofs * T_ + t) * U_ + ncol], 0, 0);
        __builtin_prefetch(&out[((long)(mofs + 7) * T_ + t) * U_ + ncol], 0, 0);

        v8f acc0 = {}, acc1 = {};
#pragma unroll 4
        for (int kk = 0; kk < U_ / 32; kk += 2) {
            V16 a0, b0, a1, b1;
            a0.h[0] = *(const v8h*)(aRow + kk * 32);
            a0.h[1] = *(const v8h*)(aRow + kk * 32 + 8);
            const _Float16* bp0 = bFrag + (size_t)kk * 1024;
            b0.h[0] = *(const v8h*)(bp0);
            b0.h[1] = *(const v8h*)(bp0 + 8);
            a1.h[0] = *(const v8h*)(aRow + kk * 32 + 32);
            a1.h[1] = *(const v8h*)(aRow + kk * 32 + 40);
            const _Float16* bp1 = bp0 + 1024;
            b1.h[0] = *(const v8h*)(bp1);
            b1.h[1] = *(const v8h*)(bp1 + 8);
            acc0 = wmma_f16(a0, b0, acc0);
            acc1 = wmma_f16(a1, b1, acc1);
        }
        v8f acc = acc0 + acc1;

#pragma unroll
        for (int e = 0; e < 8; ++e) {
            int  b_  = mofs + e;
            long off = ((long)b_ * T_ + t) * U_ + ncol;
            float v  = tanhf(acc[e] + out[off]);
            out[off] = v;
            hn[b_ * U_ + ncol] = (_Float16)v;
        }

        __threadfence();
        __syncthreads();
        if (tid == 0) {
            __hip_atomic_fetch_add(counter, 1u, __ATOMIC_RELEASE,
                                   __HIP_MEMORY_SCOPE_AGENT);
            const unsigned target = nwg * (unsigned)(t + 1);
            while (__hip_atomic_load(counter, __ATOMIC_ACQUIRE,
                                     __HIP_MEMORY_SCOPE_AGENT) < target)
                __builtin_amdgcn_s_sleep(1);
        }
        __syncthreads();
    }
}

// ---------------------------------------------------------------------------
extern "C" void kernel_launch(void* const* d_in, const int* in_sizes, int n_in,
                              void* d_out, int out_size, void* d_ws, size_t ws_size,
                              hipStream_t stream)
{
    (void)in_sizes; (void)n_in; (void)out_size; (void)ws_size;
    const float* x    = (const float*)d_in[0];   // [B,T,D]
    const float* win  = (const float*)d_in[1];   // [D,U]
    const float* wres = (const float*)d_in[2];   // [U,U]
    const float* bias = (const float*)d_in[3];   // [U]
    float* out = (float*)d_out;                  // [B,T,U]

    char* ws = (char*)d_ws;
    _Float16* a16    = (_Float16*)(ws);                         // 8 MB
    _Float16* bt16   = (_Float16*)(ws + (8u  << 20));           // 512 KB
    _Float16* wres16 = (_Float16*)(ws + (8u  << 20) + (512u << 10));   // 2 MB
    _Float16* h      = (_Float16*)(ws + (10u << 20) + (512u << 10));   // 128 KB
    unsigned* counter = (unsigned*)(ws + (10u << 20) + (704u << 10));

    esn_prep<<<2048, 256, 0, stream>>>(x, win, wres, a16, bt16, wres16, h, counter);
    esn_xin<<<dim3(64, 128), 256, 0, stream>>>(a16, bt16, bias, out);
    esn_scan<<<32, 128, 0, stream>>>(wres16, out, h, counter);
}
